// WindowAttention_19086834663771
// MI455X (gfx1250) — matrix operand order repaired and checked by reading, not compiled, not run on hardware
//
#include <hip/hip_runtime.h>

// ---------------------------------------------------------------------------
// Fused Swin window attention for gfx1250 (MI455X): QKV + attn + proj + SE
// in a single kernel. bf16 WMMA (v_wmma_f32_16x16x32_bf16), fp32 accumulate.
// B=4096 windows, N=64 tokens, C=256, H=8 heads, d=32, nW=1024.
// Weights are pre-packed once (bf16, B-fragment lane order) into d_ws.
// ---------------------------------------------------------------------------

typedef __attribute__((ext_vector_type(8)))  float          v8f;
typedef __attribute__((ext_vector_type(16))) __bf16         v16bf;
typedef __attribute__((ext_vector_type(16))) unsigned short v16u;

union FragU { v16u u; v16bf b; };

__device__ __forceinline__ unsigned short f2bf(float f) {
  unsigned int u = __float_as_uint(f);
  u += 0x7FFFu + ((u >> 16) & 1u);           // round-to-nearest-even
  return (unsigned short)(u >> 16);
}

__device__ __forceinline__ v8f vzero8() {
  v8f z;
#pragma unroll
  for (int i = 0; i < 8; ++i) z[i] = 0.0f;
  return z;
}

// A fragment (16x32 bf16): A(m,k) = s[(row0+m)*ld + k0 + k]
// lanes 0-15: M=0..15, K in {0..15}; lanes 16-31: K in {16..31}
__device__ __forceinline__ v16bf load_a(const unsigned short* s, int row0, int k0,
                                        int ld, int lane) {
  const int m = lane & 15, hs = lane >> 4;
  const unsigned short* p = s + (row0 + m) * ld + k0 + hs * 16;
  FragU f;
#pragma unroll
  for (int i = 0; i < 16; ++i) f.u[i] = p[i];
  return f.b;
}

// B fragment from pre-packed weights: one 32-byte vector load per lane.
// Packed order: [tile][lane][i], element (K = 16*(lane>>4)+i, N = lane&15).
__device__ __forceinline__ v16bf load_b_packed(const unsigned short* wp, int tile,
                                               int lane) {
  const v16u* p = (const v16u*)(wp + ((size_t)tile * 32 + lane) * 16);
  FragU f;
  f.u = *p;
  return f.b;
}

// B fragment of a transposed operand in LDS: B(k,n) = s[(n0+n)*ld + k0 + k]
// (contiguous 16 bf16 per lane -> ds_load_b128 pair when 16B-aligned)
__device__ __forceinline__ v16bf load_bT(const unsigned short* s, int n0, int k0,
                                         int ld, int lane) {
  const int n = lane & 15, hs = lane >> 4;
  const unsigned short* p = s + (n0 + n) * ld + k0 + hs * 16;
  FragU f;
#pragma unroll
  for (int i = 0; i < 16; ++i) f.u[i] = p[i];
  return f.b;
}

#define WMMA_BF16(A, B, C) \
  __builtin_amdgcn_wmma_f32_16x16x32_bf16(false, (A), false, (B), (short)0, (C), false, false)

__device__ __forceinline__ float rmax16(float v) {
#pragma unroll
  for (int m = 1; m < 16; m <<= 1) v = fmaxf(v, __shfl_xor(v, m, 32));
  return v;
}
__device__ __forceinline__ float rsum16(float v) {
#pragma unroll
  for (int m = 1; m < 16; m <<= 1) v += __shfl_xor(v, m, 32);
  return v;
}

// ---------------------------------------------------------------------------
// One-shot weight pack: fp32 -> bf16, swizzled to B-fragment lane order.
// ws layout (ushorts): [0,196608) QKV (48 ntiles x 8 ktiles x 32 lanes x 16)
//                      [196608,262144) proj (16 x 8 x 32 x 16)
// then fp32 bcat[768] = {bq | bkv}.
// ---------------------------------------------------------------------------
__global__ __launch_bounds__(256) void pack_weights_kernel(
    const float* __restrict__ Wq, const float* __restrict__ Wkv,
    const float* __restrict__ Wproj, const float* __restrict__ bq,
    const float* __restrict__ bkv, unsigned short* __restrict__ wp,
    float* __restrict__ bcat) {
  const int tid = blockIdx.x * 256 + threadIdx.x;
  if (tid < 196608) {                       // QKV: 768 output cols, K=256
    const int i = tid & 15, lane = (tid >> 4) & 31;
    const int kk = (tid >> 9) & 7, nt = tid >> 12;
    const int k = kk * 32 + (lane >> 4) * 16 + i;
    const int n = nt * 16 + (lane & 15);
    const float v = (n < 256) ? Wq[k * 256 + n] : Wkv[k * 512 + (n - 256)];
    wp[tid] = f2bf(v);
  } else if (tid < 262144) {                // proj: 256 cols, K=256
    const int p = tid - 196608;
    const int i = p & 15, lane = (p >> 4) & 31;
    const int kk = (p >> 9) & 7, nt = p >> 12;
    const int k = kk * 32 + (lane >> 4) * 16 + i;
    const int n = nt * 16 + (lane & 15);
    wp[tid] = f2bf(Wproj[k * 256 + n]);
  } else if (tid < 262912) {                // concatenated biases
    const int c = tid - 262144;
    bcat[c] = (c < 256) ? bq[c] : bkv[c - 256];
  }
}

__global__ __launch_bounds__(256) void swin_fused_kernel(
    const float* __restrict__ x, const float* __restrict__ mask,
    const unsigned short* __restrict__ wqkv_p,
    const unsigned short* __restrict__ wproj_p,
    const float* __restrict__ bcat, const float* __restrict__ bias_table,
    const float* __restrict__ bproj, const float* __restrict__ Wse1,
    const float* __restrict__ Wse2, float* __restrict__ out) {
  constexpr int N = 64, C = 256, LDX = 264, LDP = 72, LDV = 72;
  extern __shared__ unsigned short sm[];
  unsigned short* xs  = sm;                  // 64 x LDX bf16 (x; reused as attn-out)
  unsigned short* qs  = xs + N * LDX;        // 64 x LDX bf16 (scaled q)
  unsigned short* ks  = qs + N * LDX;        // 64 x LDX bf16 (k, row-major)
  unsigned short* vsT = ks + N * LDX;        // 256 x LDV bf16 (v transposed)
  unsigned short* ps  = vsT + C * LDV;       // 8 waves x 16 x LDP bf16 (softmax P)
  float* ycol = (float*)(ps + 8 * 16 * LDP); // 256: token-mean of proj out
  float* zse  = ycol + C;                    // 16 : SE hidden
  float* sse  = zse + 16;                    // 256: SE gate

  const int tid   = threadIdx.x;
  const int wave  = tid >> 5;
  const int lane  = tid & 31;
  const int nlane = lane & 15;
  const int hs    = lane >> 4;
  const size_t b  = blockIdx.x;

  // ---- stage 1: x_b [64,256] fp32 -> bf16 LDS ----
  const float4* xb4 = (const float4*)(x + b * (size_t)(N * C));
  for (int i = tid; i < (N * C) / 4; i += 256) {
    float4 v = xb4[i];
    int base = i * 4, r = base >> 8, c = base & 255;
    unsigned short* d = xs + r * LDX + c;
    d[0] = f2bf(v.x); d[1] = f2bf(v.y); d[2] = f2bf(v.z); d[3] = f2bf(v.w);
  }
  __syncthreads();

  // ---- stage 2: QKV = x_b @ [Wq | Wkv] + bias; q scaled by d^-0.5 ----
  // 48 N-tiles over 768 output cols; wave owns 6 N-tiles x 4 M-tiles
#pragma unroll 1
  for (int ntl = 0; ntl < 6; ++ntl) {
    const int ntile = wave * 6 + ntl;
    const int ncol0 = ntile * 16;
    v8f acc[4];
#pragma unroll
    for (int mt = 0; mt < 4; ++mt) acc[mt] = vzero8();
#pragma unroll
    for (int kk = 0; kk < 8; ++kk) {
      v16bf bf = load_b_packed(wqkv_p, ntile * 8 + kk, lane);
#pragma unroll
      for (int mt = 0; mt < 4; ++mt) {
        v16bf af = load_a(xs, mt * 16, kk * 32, LDX, lane);
        acc[mt] = WMMA_BF16(af, bf, acc[mt]);
      }
    }
    const int col = ncol0 + nlane;
    const float bias = bcat[col];
    if (ncol0 < 256) {                       // q tile (wave-uniform branch)
#pragma unroll
      for (int mt = 0; mt < 4; ++mt)
#pragma unroll
        for (int g = 0; g < 8; ++g)
          qs[(mt * 16 + g + 8 * hs) * LDX + col] =
              f2bf((acc[mt][g] + bias) * 0.17677669529663687f);
    } else if (ncol0 < 512) {                // k tile
#pragma unroll
      for (int mt = 0; mt < 4; ++mt)
#pragma unroll
        for (int g = 0; g < 8; ++g)
          ks[(mt * 16 + g + 8 * hs) * LDX + (col - 256)] = f2bf(acc[mt][g] + bias);
    } else {                                 // v tile, stored transposed
#pragma unroll
      for (int mt = 0; mt < 4; ++mt)
#pragma unroll
        for (int g = 0; g < 8; ++g)
          vsT[(col - 512) * LDV + (mt * 16 + g + 8 * hs)] = f2bf(acc[mt][g] + bias);
    }
  }
  __syncthreads();

  // ---- stage 3: per-head attention (wave == head) ----
  const int h = wave;
  const float* maskw = mask + (size_t)(b & 1023) * (64 * 64);
  unsigned short* psw = ps + wave * 16 * LDP;
  unsigned short* os  = xs;  // reuse x region for attention output (bf16)
#pragma unroll 1
  for (int mb = 0; mb < 4; ++mb) {
    v8f s[4];
    v16bf aq = load_a(qs, mb * 16, h * 32, LDX, lane);  // q rows, K = head dim
#pragma unroll
    for (int nt = 0; nt < 4; ++nt) {
      v16bf bk = load_bT(ks, nt * 16, h * 32, LDX, lane);  // B(k,n) = k[n][k]
      s[nt] = WMMA_BF16(aq, bk, vzero8());
    }
    // relative-position bias + shifted-window mask
#pragma unroll
    for (int nt = 0; nt < 4; ++nt) {
      const int n = nt * 16 + nlane, yj = n >> 3, xj = n & 7;
#pragma unroll
      for (int g = 0; g < 8; ++g) {
        const int m = mb * 16 + g + 8 * hs;
        const int ridx = ((m >> 3) - yj + 7) * 15 + ((m & 7) - xj + 7);
        s[nt][g] += bias_table[ridx * 8 + h] + maskw[m * 64 + n];
      }
    }
    // row softmax (row lives on a 16-lane group x 4 tiles at fixed g)
#pragma unroll
    for (int g = 0; g < 8; ++g) {
      float mx = fmaxf(fmaxf(s[0][g], s[1][g]), fmaxf(s[2][g], s[3][g]));
      mx = rmax16(mx);
      float psum = 0.0f;
#pragma unroll
      for (int nt = 0; nt < 4; ++nt) { s[nt][g] = __expf(s[nt][g] - mx); psum += s[nt][g]; }
      const float inv = 1.0f / rsum16(psum);
#pragma unroll
      for (int nt = 0; nt < 4; ++nt)
        psw[(g + 8 * hs) * LDP + nt * 16 + nlane] = f2bf(s[nt][g] * inv);
    }
    // out_h[16,32] = P[16,64] @ v_h[64,32]
#pragma unroll
    for (int vt = 0; vt < 2; ++vt) {
      v8f o = vzero8();
#pragma unroll
      for (int k2 = 0; k2 < 2; ++k2) {
        v16bf ap = load_a(psw, 0, k2 * 32, LDP, lane);
        v16bf bv = load_bT(vsT, h * 32 + vt * 16, k2 * 32, LDV, lane);
        o = WMMA_BF16(ap, bv, o);
      }
#pragma unroll
      for (int g = 0; g < 8; ++g)
        os[(mb * 16 + g + 8 * hs) * LDX + h * 32 + vt * 16 + nlane] = f2bf(o[g]);
    }
  }
  __syncthreads();

  // ---- stage 4: proj = os @ Wproj + bproj; column means for SE ----
  v8f po[2][4];
#pragma unroll 1
  for (int ntl = 0; ntl < 2; ++ntl) {
    const int ntile = wave * 2 + ntl;
    const int ncol0 = ntile * 16;
#pragma unroll
    for (int mt = 0; mt < 4; ++mt) po[ntl][mt] = vzero8();
#pragma unroll
    for (int kk = 0; kk < 8; ++kk) {
      v16bf bf = load_b_packed(wproj_p, ntile * 8 + kk, lane);
#pragma unroll
      for (int mt = 0; mt < 4; ++mt) {
        v16bf af = load_a(os, mt * 16, kk * 32, LDX, lane);
        po[ntl][mt] = WMMA_BF16(af, bf, po[ntl][mt]);
      }
    }
    const int col = ncol0 + nlane;
    const float bb = bproj[col];
    float csum = 0.0f;
#pragma unroll
    for (int mt = 0; mt < 4; ++mt)
#pragma unroll
      for (int g = 0; g < 8; ++g) { po[ntl][mt][g] += bb; csum += po[ntl][mt][g]; }
    csum += __shfl_xor(csum, 16, 32);        // join both 32-row halves
    if (hs == 0) ycol[col] = csum * (1.0f / 64.0f);
  }
  __syncthreads();

  // ---- stage 5: SE gate (block-local; this block owns all 64 tokens) ----
  if (tid < 16) {
    float z = 0.0f;
    for (int c = 0; c < 256; ++c) z += ycol[c] * Wse1[c * 16 + tid];
    zse[tid] = fmaxf(z, 0.0f);
  }
  __syncthreads();
  {
    float sacc = 0.0f;
#pragma unroll
    for (int j = 0; j < 16; ++j) sacc += zse[j] * Wse2[j * 256 + tid];
    sse[tid] = 1.0f / (1.0f + __expf(-sacc));
  }
  __syncthreads();

  // ---- stage 6: gate + store fp32 output ----
  float* ob = out + b * (size_t)(N * C);
#pragma unroll
  for (int ntl = 0; ntl < 2; ++ntl) {
    const int col = (wave * 2 + ntl) * 16 + nlane;
    const float sc = sse[col];
#pragma unroll
    for (int mt = 0; mt < 4; ++mt)
#pragma unroll
      for (int g = 0; g < 8; ++g)
        ob[(mt * 16 + g + 8 * hs) * 256 + col] = po[ntl][mt][g] * sc;
  }
}

extern "C" void kernel_launch(void* const* d_in, const int* in_sizes, int n_in,
                              void* d_out, int out_size, void* d_ws, size_t ws_size,
                              hipStream_t stream) {
  (void)in_sizes; (void)n_in; (void)ws_size; (void)out_size;
  const float* x          = (const float*)d_in[0];
  const float* mask       = (const float*)d_in[1];
  const float* Wq         = (const float*)d_in[2];
  const float* bq         = (const float*)d_in[3];
  const float* Wkv        = (const float*)d_in[4];
  const float* bkv        = (const float*)d_in[5];
  const float* bias_table = (const float*)d_in[6];
  const float* Wproj      = (const float*)d_in[7];
  const float* bproj      = (const float*)d_in[8];
  const float* Wse1       = (const float*)d_in[9];
  const float* Wse2       = (const float*)d_in[10];

  unsigned short* wp   = (unsigned short*)d_ws;      // 262144 bf16 packed weights
  float*          bcat = (float*)(wp + 262144);      // 768 fp32 concat biases

  pack_weights_kernel<<<(262912 + 255) / 256, 256, 0, stream>>>(
      Wq, Wkv, Wproj, bq, bkv, wp, bcat);

  constexpr size_t SHMEM =
      (size_t)(3 * 64 * 264 + 256 * 72 + 8 * 16 * 72) * sizeof(unsigned short) +
      (size_t)(256 + 16 + 256) * sizeof(float);      // ~155 KB (2 WGs / 320KB WGP)

  hipFuncSetAttribute((const void*)swin_fused_kernel,
                      hipFuncAttributeMaxDynamicSharedMemorySize, (int)SHMEM);

  swin_fused_kernel<<<4096, 256, SHMEM, stream>>>(
      x, mask, wp, wp + 196608, bcat, bias_table, bproj, Wse1, Wse2,
      (float*)d_out);
}
